// DeepseekMoE_32495722561858
// MI455X (gfx1250) — matrix operand order repaired and checked by reading, not compiled, not run on hardware
//
#include <hip/hip_runtime.h>

// ---------------- problem constants ----------------
#define TT   512          // tokens
#define HH   2048         // hidden
#define EE   32           // routed experts
#define GG   8            // groups
#define TKGc 4            // topk groups
#define KKc  6            // top-k experts
#define II   1408         // moe intermediate
#define NSI  2816         // shared intermediate (NS*I)
#define SCALE_F 2.5f

typedef __attribute__((ext_vector_type(16))) __bf16 v16bf;
typedef __attribute__((ext_vector_type(2)))  __bf16 v2bf;
typedef __attribute__((ext_vector_type(8)))  float  v8f;
typedef __attribute__((ext_vector_type(4)))  float  v4f;

__device__ __forceinline__ v8f wmma_bf16(v16bf a, v16bf b, v8f c) {
  // D = A(16x32 bf16) x B(32x16 bf16) + C(16x16 f32)
  return __builtin_amdgcn_wmma_f32_16x16x32_bf16(false, a, false, b, (short)0, c,
                                                 false, false);
}

__device__ __forceinline__ float fast_silu(float g) {
  // g * sigmoid(g) with v_rcp_f32 instead of IEEE divide chain
  return g * __builtin_amdgcn_rcpf(1.f + __expf(-g));
}

// ---------------------------------------------------------------------------
// Kernel 1: router gate + grouped top-k routing + bf16 copy of x.
// One block per token. 256 threads: 32 experts x 8 K-chunks for the gate dot.
// ---------------------------------------------------------------------------
__global__ void k_route(const float* __restrict__ x, const float* __restrict__ gw,
                        const float* __restrict__ bias,
                        int* __restrict__ cnt, int* __restrict__ tokbuf,
                        float* __restrict__ wtbuf, int* __restrict__ topk_e,
                        int* __restrict__ topk_pos, __bf16* __restrict__ xb) {
  const int t = blockIdx.x;
  const int tid = threadIdx.x;
  __shared__ float xs[HH];       // 8KB token row
  __shared__ float red[256];
  __shared__ float sc[EE];

  for (int i = tid * 4; i < HH; i += 256 * 4) {
    v4f v = *(const v4f*)(x + (size_t)t * HH + i);
    *(v4f*)&xs[i] = v;
    v2bf p0; p0[0] = (__bf16)v[0]; p0[1] = (__bf16)v[1];
    v2bf p1; p1[0] = (__bf16)v[2]; p1[1] = (__bf16)v[3];
    *(v2bf*)(xb + (size_t)t * HH + i)     = p0;   // bf16 copy for WMMA
    *(v2bf*)(xb + (size_t)t * HH + i + 2) = p1;
  }
  __syncthreads();

  const int e = tid & 31, ch = tid >> 5;  // expert, K-chunk
  float p = 0.f;
  const int kbase = ch * 256;
  for (int j = 0; j < 256; ++j) {
    p += xs[kbase + j] * gw[(size_t)(kbase + j) * EE + e];  // coalesced over e
  }
  red[tid] = p;
  __syncthreads();
  if (tid < EE) {
    float s = 0.f;
    for (int c = 0; c < 8; ++c) s += red[c * 32 + tid];
    sc[tid] = 1.f / (1.f + __expf(-s));   // sigmoid score (precise path)
  }
  __syncthreads();

  if (tid == 0) {
    const float NEG = -3.0e38f;
    float sfc[EE];
    for (int i = 0; i < EE; ++i) sfc[i] = sc[i] + bias[i];
    // group score = sum of top-2 of each group of 4
    float gs[GG];
    for (int g = 0; g < GG; ++g) {
      float m1 = NEG, m2 = NEG;
      for (int j = 0; j < 4; ++j) {
        float v = sfc[g * 4 + j];
        if (v > m1) { m2 = m1; m1 = v; }
        else if (v > m2) { m2 = v; }
      }
      gs[g] = m1 + m2;
    }
    // select top TKG groups
    bool gsel[GG];
    for (int g = 0; g < GG; ++g) gsel[g] = false;
    for (int it = 0; it < TKGc; ++it) {
      float best = NEG; int bi = 0;
      for (int g = 0; g < GG; ++g)
        if (!gsel[g] && gs[g] > best) { best = gs[g]; bi = g; }
      gsel[bi] = true;
    }
    // top-K experts among selected groups (bias-corrected score for selection)
    float masked[EE];
    for (int i = 0; i < EE; ++i) masked[i] = gsel[i >> 2] ? sfc[i] : NEG;
    int ids[KKc]; float tw[KKc]; float sum = 0.f;
    for (int kk = 0; kk < KKc; ++kk) {
      float best = NEG; int bi = 0;
      for (int i = 0; i < EE; ++i)
        if (masked[i] > best) { best = masked[i]; bi = i; }
      masked[bi] = NEG;
      ids[kk] = bi;
      tw[kk] = sc[bi];                 // weight from raw sigmoid score
      sum += tw[kk];
    }
    for (int kk = 0; kk < KKc; ++kk) {
      float w = tw[kk] / sum * SCALE_F;
      int ee = ids[kk];
      int pos = atomicAdd(&cnt[ee], 1);
      tokbuf[ee * TT + pos] = t;
      wtbuf[ee * TT + pos] = w;
      topk_e[t * KKc + kk] = ee;
      topk_pos[t * KKc + kk] = pos;
    }
  }
}

// ---------------------------------------------------------------------------
// Kernel 2: exclusive prefix sum of expert counts (E=32, trivial)
// ---------------------------------------------------------------------------
__global__ void k_offs(const int* __restrict__ cnt, int* __restrict__ offs) {
  if (threadIdx.x == 0 && blockIdx.x == 0) {
    int a = 0;
    for (int e = 0; e < EE; ++e) { offs[e] = a; a += cnt[e]; }
  }
}

// ---------------------------------------------------------------------------
// GEMM1 + SiLU*mul.  Tile: 16(M) x 128(act N); 8 waves, each wave owns one
// 16x16 act tile (gate + up accumulators). K chunked by 32 (one WMMA depth).
// ROUTED: gathered token rows, per-row combine weight folded into act.
// ---------------------------------------------------------------------------
template <bool ROUTED>
__global__ void k_gemm1(const __bf16* __restrict__ xb, const float* __restrict__ W,
                        int ldw, int upOff,
                        const int* __restrict__ cnt, const int* __restrict__ offs,
                        const int* __restrict__ tokbuf, const float* __restrict__ wtbuf,
                        __bf16* __restrict__ actOut, int ldact) {
  const int e  = ROUTED ? blockIdx.z : 0;
  const int ne = ROUTED ? cnt[e] : TT;
  const int m0 = blockIdx.y * 16;
  if (m0 >= ne) return;
  const int n0 = blockIdx.x * 128;

  __shared__ __bf16 Al[16 * 32];    // A chunk (tokens x K)
  __shared__ __bf16 Bg[128 * 32];   // gate W chunk, transposed [n][k]
  __shared__ __bf16 Bu[128 * 32];   // up   W chunk, transposed [n][k]
  __shared__ float  rs[16];
  __shared__ int    rtok[16];

  const int tid = threadIdx.x;
  if (tid < 16) {
    int m = m0 + tid;
    if (ROUTED) {
      bool v = m < ne;
      rtok[tid] = v ? tokbuf[e * TT + m] : tokbuf[e * TT];
      rs[tid]   = v ? wtbuf[e * TT + m] : 0.f;
    } else {
      rtok[tid] = m;
      rs[tid]   = 1.f;
    }
  }

  const float* wb = W + (size_t)e * HH * (size_t)ldw;
  const int lane = tid & 31, wv = tid >> 5;
  v8f cg = {}; v8f cu = {};

  for (int k0 = 0; k0 < HH; k0 += 32) {
    __syncthreads();
    // stage A: 16x32 bf16 = 256 k-pairs, one u32 copy per thread
    {
      int m = tid >> 4, kp = tid & 15;
      unsigned v = *(const unsigned*)(xb + (size_t)rtok[m] * HH + k0 + 2 * kp);
      *(unsigned*)&Al[m * 32 + 2 * kp] = v;
    }
    // stage B gate+up: 32x128 f32 each -> bf16 [n][k], k-pairs packed (b32)
#pragma unroll
    for (int it = 0; it < 2; ++it) {
      int idx = it * 256 + tid;                      // 0..511
      int kp  = (idx >> 2) & 15;                     // k-pair (k = 2*kp)
      int c4  = ((idx & 3) | ((idx >> 6) << 2)) * 4; // 0,4,...,124
      const float* r0 = wb + (size_t)(k0 + 2 * kp) * ldw + n0 + c4;
      const float* r1 = r0 + ldw;
      v4f g0 = *(const v4f*)r0;
      v4f g1 = *(const v4f*)r1;
      v4f u0 = *(const v4f*)(r0 + upOff);
      v4f u1 = *(const v4f*)(r1 + upOff);
      if (it == 0 && k0 + 32 < HH)
        __builtin_prefetch((const void*)(r0 + 32 * (size_t)ldw), 0, 3);
#pragma unroll
      for (int j = 0; j < 4; ++j) {
        v2bf pg; pg[0] = (__bf16)g0[j]; pg[1] = (__bf16)g1[j];
        v2bf pu; pu[0] = (__bf16)u0[j]; pu[1] = (__bf16)u1[j];
        *(v2bf*)&Bg[(c4 + j) * 32 + 2 * kp] = pg;
        *(v2bf*)&Bu[(c4 + j) * 32 + 2 * kp] = pu;
      }
    }
    __syncthreads();

    // A frag: lanes0-15 K0-15, lanes16-31 K16-31 (same k order as B)
    const int mm = lane & 15, hf = lane >> 4;
    v16bf a, bg, bu;
    const __bf16* pa = &Al[mm * 32 + 16 * hf];
#pragma unroll
    for (int j = 0; j < 8; ++j) { a[j] = pa[j]; a[8 + j] = pa[8 + j]; }
    // B frag: lanes0-15 K0-15, lanes16-31 K16-31, n = lane&15 (same k order as A)
    const __bf16* pg = &Bg[(wv * 16 + mm) * 32 + 16 * hf];
    const __bf16* pu = &Bu[(wv * 16 + mm) * 32 + 16 * hf];
#pragma unroll
    for (int j = 0; j < 16; ++j) { bg[j] = pg[j]; bu[j] = pu[j]; }
    cg = wmma_bf16(a, bg, cg);
    cu = wmma_bf16(a, bu, cu);
  }
  __syncthreads();

  const int hf = lane >> 4, col = lane & 15;
#pragma unroll
  for (int r = 0; r < 8; ++r) {
    int row = r + 8 * hf;            // C layout: VGPR r -> M=r / M=r+8
    int m = m0 + row;
    if (m < ne) {
      float g = (float)cg[r], u = (float)cu[r];
      float av = fast_silu(g) * u * rs[row];   // combine weight folded (ROUTED)
      size_t orow = ROUTED ? (size_t)(offs[e] + m) : (size_t)m;
      actOut[orow * (size_t)ldact + n0 + wv * 16 + col] = (__bf16)av;
    }
  }
}

// ---------------------------------------------------------------------------
// GEMM2: act(bf16) @ W2(f32->bf16) -> f32 rows. Same tiling as GEMM1.
// ---------------------------------------------------------------------------
template <bool ROUTED>
__global__ void k_gemm2(const __bf16* __restrict__ A, const float* __restrict__ W,
                        int Kdim, const int* __restrict__ cnt,
                        const int* __restrict__ offs, float* __restrict__ outB) {
  const int e  = ROUTED ? blockIdx.z : 0;
  const int ne = ROUTED ? cnt[e] : TT;
  const int m0 = blockIdx.y * 16;
  if (m0 >= ne) return;
  const int n0 = blockIdx.x * 128;
  const int rbase = ROUTED ? offs[e] : 0;

  __shared__ __bf16 Al[16 * 32];
  __shared__ __bf16 Bl[128 * 32];
  __shared__ int    rrow[16];

  const int tid = threadIdx.x;
  const int lane = tid & 31, wv = tid >> 5;
  if (tid < 16) {
    int mr = m0 + tid; if (mr >= ne) mr = ne - 1;   // clamp (masked at store)
    rrow[tid] = rbase + mr;
  }
  const float* wb = W + (size_t)e * (size_t)Kdim * HH;
  v8f cc = {};

  for (int k0 = 0; k0 < Kdim; k0 += 32) {
    __syncthreads();
    // stage A: one u32 (bf16 pair) per thread
    {
      int m = tid >> 4, kp = tid & 15;
      unsigned v = *(const unsigned*)(A + (size_t)rrow[m] * Kdim + k0 + 2 * kp);
      *(unsigned*)&Al[m * 32 + 2 * kp] = v;
    }
    // stage B: 32x128 f32 -> bf16 [n][k], k-pairs packed
#pragma unroll
    for (int it = 0; it < 2; ++it) {
      int idx = it * 256 + tid;
      int kp  = (idx >> 2) & 15;
      int c4  = ((idx & 3) | ((idx >> 6) << 2)) * 4;
      const float* r0 = wb + (size_t)(k0 + 2 * kp) * HH + n0 + c4;
      const float* r1 = r0 + HH;
      v4f b0 = *(const v4f*)r0;
      v4f b1 = *(const v4f*)r1;
      if (it == 0 && k0 + 32 < Kdim)
        __builtin_prefetch((const void*)(r0 + 32 * (size_t)HH), 0, 3);
#pragma unroll
      for (int j = 0; j < 4; ++j) {
        v2bf pb; pb[0] = (__bf16)b0[j]; pb[1] = (__bf16)b1[j];
        *(v2bf*)&Bl[(c4 + j) * 32 + 2 * kp] = pb;
      }
    }
    __syncthreads();

    const int mm = lane & 15, hf = lane >> 4;
    v16bf a, b;
    const __bf16* pa = &Al[mm * 32 + 16 * hf];
#pragma unroll
    for (int j = 0; j < 8; ++j) { a[j] = pa[j]; a[8 + j] = pa[8 + j]; }
    const __bf16* pb = &Bl[(wv * 16 + mm) * 32 + 16 * hf];
#pragma unroll
    for (int j = 0; j < 16; ++j) b[j] = pb[j];
    cc = wmma_bf16(a, b, cc);
  }

  const int hf = lane >> 4, col = lane & 15;
#pragma unroll
  for (int r = 0; r < 8; ++r) {
    int row = r + 8 * hf;
    int m = m0 + row;
    if (m < ne) {
      outB[(size_t)(rbase + m) * HH + n0 + wv * 16 + col] = (float)cc[r];
    }
  }
}

// ---------------------------------------------------------------------------
// Gather: out[t] = shared[t] (already in d_out) + sum_k eo[slot(t,k)]
// Fixed k-order summation -> bitwise deterministic. b128 vectorized.
// ---------------------------------------------------------------------------
__global__ void k_gather(float* __restrict__ out, const float* __restrict__ eo,
                         const int* __restrict__ offs, const int* __restrict__ topk_e,
                         const int* __restrict__ topk_pos) {
  const int t = blockIdx.x, tid = threadIdx.x;
  int slot[KKc];
#pragma unroll
  for (int k = 0; k < KKc; ++k)
    slot[k] = offs[topk_e[t * KKc + k]] + topk_pos[t * KKc + k];
#pragma unroll
  for (int hv = 0; hv < 2; ++hv) {
    int h = (hv * 256 + tid) * 4;
    v4f acc = *(const v4f*)(out + (size_t)t * HH + h);
#pragma unroll
    for (int k = 0; k < KKc; ++k)
      acc += *(const v4f*)(eo + (size_t)slot[k] * HH + h);
    *(v4f*)(out + (size_t)t * HH + h) = acc;
  }
}

// ---------------------------------------------------------------------------
extern "C" void kernel_launch(void* const* d_in, const int* in_sizes, int n_in,
                              void* d_out, int out_size, void* d_ws, size_t ws_size,
                              hipStream_t stream) {
  const float* x       = (const float*)d_in[0];   // hidden_states [T,H]
  // d_in[1] = residual (unused by reference output)
  const float* gate_w  = (const float*)d_in[2];   // [H,E]
  const float* bias    = (const float*)d_in[3];   // [E]
  const float* w13     = (const float*)d_in[4];   // [E,H,2I]
  const float* w2      = (const float*)d_in[5];   // [E,I,H]
  const float* sh_gu   = (const float*)d_in[6];   // [H,2*NS*I]
  const float* sh_dn   = (const float*)d_in[7];   // [NS*I,H]
  float* out = (float*)d_out;                     // [T,H]

  // ---- workspace layout (bytes) ----
  char* w = (char*)d_ws;
  constexpr size_t OFF_CNT = 0;                                  // E*4
  constexpr size_t OFF_OFF = 256;                                // E*4
  constexpr size_t OFF_TOK = 512;                                // E*T*4
  constexpr size_t OFF_WT  = OFF_TOK + (size_t)EE * TT * 4;      // E*T*4
  constexpr size_t OFF_TE  = OFF_WT  + (size_t)EE * TT * 4;      // T*K*4
  constexpr size_t OFF_TP  = OFF_TE  + (size_t)TT * KKc * 4;     // T*K*4
  constexpr size_t OFF_XB  = OFF_TP  + (size_t)TT * KKc * 4;     // T*H*2
  constexpr size_t OFF_ACT = OFF_XB  + (size_t)TT * HH * 2;      // (T*K)*I*2
  constexpr size_t OFF_SHA = OFF_ACT + (size_t)TT * KKc * II * 2; // T*NSI*2
  constexpr size_t OFF_EO  = OFF_SHA + (size_t)TT * NSI * 2;     // (T*K)*H*4
  int*    cnt      = (int*)(w + OFF_CNT);
  int*    offs     = (int*)(w + OFF_OFF);
  int*    tokbuf   = (int*)(w + OFF_TOK);
  float*  wtbuf    = (float*)(w + OFF_WT);
  int*    topk_e   = (int*)(w + OFF_TE);
  int*    topk_pos = (int*)(w + OFF_TP);
  __bf16* xb       = (__bf16*)(w + OFF_XB);
  __bf16* act      = (__bf16*)(w + OFF_ACT);
  __bf16* shact    = (__bf16*)(w + OFF_SHA);
  float*  eo       = (float*)(w + OFF_EO);

  hipMemsetAsync(cnt, 0, EE * sizeof(int), stream);

  k_route<<<TT, 256, 0, stream>>>(x, gate_w, bias, cnt, tokbuf, wtbuf,
                                  topk_e, topk_pos, xb);
  k_offs<<<1, 64, 0, stream>>>(cnt, offs);

  // routed GEMM1: act N = 1408 -> 11 tiles of 128
  k_gemm1<true><<<dim3(II / 128, TT / 16, EE), 256, 0, stream>>>(
      xb, w13, 2 * II, II, cnt, offs, tokbuf, wtbuf, act, II);
  // shared GEMM1: act N = 2816 -> 22 tiles
  k_gemm1<false><<<dim3(NSI / 128, TT / 16, 1), 256, 0, stream>>>(
      xb, sh_gu, 2 * NSI, NSI, nullptr, nullptr, nullptr, nullptr, shact, NSI);

  // routed GEMM2: K = 1408, N = 2048 -> eo rows (weight already folded)
  k_gemm2<true><<<dim3(HH / 128, TT / 16, EE), 256, 0, stream>>>(
      act, w2, II, cnt, offs, eo);
  // shared GEMM2: K = 2816, writes shared result straight into d_out
  k_gemm2<false><<<dim3(HH / 128, TT / 16, 1), 256, 0, stream>>>(
      shact, sh_dn, NSI, nullptr, nullptr, out);

  k_gather<<<TT, 256, 0, stream>>>(out, eo, offs, topk_e, topk_pos);
  (void)in_sizes; (void)n_in; (void)out_size; (void)ws_size;
}
